// TGNEncoder_28355374088724
// MI455X (gfx1250) — compile-verified, matrix operand order checked by reading, not run on hardware
//
#include <hip/hip_runtime.h>
#include <cstddef>

// ---------------- CDNA5 (gfx1250, wave32) WMMA types ----------------
typedef __attribute__((ext_vector_type(16))) __bf16 v16bf;
typedef __attribute__((ext_vector_type(8)))  __bf16 v8bf;
typedef __attribute__((ext_vector_type(8)))  float  v8f;
typedef __attribute__((ext_vector_type(4)))  int    v4i;

union BF16Frag { v16bf v; v8bf h[2]; };

struct __align__(16) F4 { float x, y, z, w; };
struct __align__(16) U4 { unsigned x, y, z, w; };

constexpr int HD  = 256;  // HID
constexpr int NH  = 4;    // HEADS
constexpr int DHD = 64;   // DH

// CDNA5 async global->LDS copy (ASYNCcnt-tracked). Guarded so both toolchains build.
#if defined(__HIP_DEVICE_COMPILE__) && __has_builtin(__builtin_amdgcn_global_load_async_to_lds_b128)
#define HAVE_ASYNC_LDS 1
#else
#define HAVE_ASYNC_LDS 0
#endif

__device__ __forceinline__ void wait_async_lds() {
#if defined(__HIP_DEVICE_COMPILE__)
#if __has_builtin(__builtin_amdgcn_s_wait_asynccnt)
  __builtin_amdgcn_s_wait_asynccnt(0);
#else
  asm volatile("s_wait_asynccnt 0x0" ::: "memory");
#endif
#endif
}

__device__ __forceinline__ float sigf(float x) { return 1.f / (1.f + __expf(-x)); }

__device__ __forceinline__ void atomicMaxF(float* addr, float val) {
  // orderable-int trick; buffer initialized to -1e30f
  if (val >= 0.f) atomicMax((int*)addr, __float_as_int(val));
  else            atomicMin((unsigned int*)addr, (unsigned int)__float_as_int(val));
}

// ==================================================================
// bf16 WMMA GEMM:  C[M,N] = act( A[M,K] @ B + bias + bias2 + beta*C )
// A is bf16 row-major (lda=K) -> staged via async global->LDS b128 copies.
// AMODE 0: A from memory. AMODE 1: A computed (Time2Vec of td/omega/phi).
// BT 0: B fp32 row-major [K,N]; BT 1: B[k][n] = Bm[n*K+k] (B stored [N,K]).
// STORE16: additionally emit bf16 copy of C (feeds downstream GEMMs).
// Block: 256 thr = 8 waves; block tile 128x64; wave tile 32x32 (2x2 of 16x16x32).
// ==================================================================
template<int AMODE, int BT, int ACT, int BETA, int STORE16>
__global__ __launch_bounds__(256)
void gemm_wmma(const __bf16* __restrict__ A, const float* __restrict__ Bm,
               const float* __restrict__ bias, const float* __restrict__ bias2,
               float* __restrict__ C, __bf16* __restrict__ C16,
               int M, int N, int K,
               const float* __restrict__ td, const float* __restrict__ omega,
               const float* __restrict__ phi)
{
  __shared__ __bf16 sA[128 * 32];   // [row][k]
  __shared__ __bf16 sB[64 * 32];    // [n][k]

  const int tid  = threadIdx.x;
  const int m0   = blockIdx.x * 128;
  const int n0   = blockIdx.y * 64;
  const int wave = tid >> 5;
  const int lane = tid & 31;
  const int wm   = wave >> 1;   // 0..3
  const int wn   = wave & 1;    // 0..1
  const int half = lane >> 4;   // 0,1
  const int lrow = lane & 15;

  // Tail M-block: pre-zero the A tile once; OOB rows are never overwritten.
  if (AMODE == 0 && m0 + 128 > M) {
    const U4 z4 = {0u, 0u, 0u, 0u};
    for (int i = tid; i < 128 * 4; i += 256) *(U4*)&sA[i * 8] = z4;
  }

  v8f acc[2][2] = {};

  for (int k0 = 0; k0 < K; k0 += 32) {
    __syncthreads();
    // ---- stage A tile (128 x 32 bf16 = 8 KB)
    if (AMODE == 0) {
#if HAVE_ASYNC_LDS
      #pragma unroll
      for (int i = tid; i < 512; i += 256) {          // 512 x 16B chunks
        const int r = i >> 2, c = (i & 3) << 3;       // 8 bf16 per 16B
        const int gr = m0 + r;
        if (gr < M)
          __builtin_amdgcn_global_load_async_to_lds_b128(
              (__attribute__((address_space(1))) v4i*)(A + (size_t)gr * K + k0 + c),
              (__attribute__((address_space(3))) v4i*)&sA[r * 32 + c], 0, 0);
      }
#else
      #pragma unroll
      for (int i = tid; i < 512; i += 256) {
        const int r = i >> 2, c = (i & 3) << 3;
        const int gr = m0 + r;
        if (gr < M) {
          *(U4*)&sA[r * 32 + c] = *(const U4*)(A + (size_t)gr * K + k0 + c);
          if (k0 + 32 < K) __builtin_prefetch(A + (size_t)gr * K + k0 + 32 + c, 0, 0);
        }
      }
#endif
    } else {
      // Time2Vec features computed in-register, stored bf16 (OOB rows -> 0)
      for (int i = tid; i < 128 * 8; i += 256) {
        const int r  = i >> 3;
        const int cg = (i & 7) << 2;
        const int gr = m0 + r;
        float tmp[4] = {0.f, 0.f, 0.f, 0.f};
        if (gr < M) {
          const float t = td[gr];
          #pragma unroll
          for (int c = 0; c < 4; ++c) {
            const int k  = k0 + cg + c;
            const float ts = t * omega[k] + phi[k];
            tmp[c] = (k == 0) ? ts : __sinf(ts);
          }
        }
        __bf16* dstp = &sA[r * 32 + cg];
        dstp[0] = (__bf16)tmp[0]; dstp[1] = (__bf16)tmp[1];
        dstp[2] = (__bf16)tmp[2]; dstp[3] = (__bf16)tmp[3];
      }
    }
    // ---- stage B tile -> sB[n][k] (fp32 weights -> bf16; small, L2-resident)
    if (BT == 0) {
      for (int i = tid; i < 64 * 32; i += 256) {
        const int n = i & 63, k = i >> 6;
        const int gn = n0 + n;
        float v = 0.f;
        if (gn < N) v = Bm[(size_t)(k0 + k) * N + gn];
        sB[n * 32 + k] = (__bf16)v;
      }
    } else {
      for (int i = tid; i < 64 * 32; i += 256) {
        const int k = i & 31, n = i >> 5;
        const int gn = n0 + n;
        float v = 0.f;
        if (gn < N) v = Bm[(size_t)gn * K + k0 + k];
        sB[n * 32 + k] = (__bf16)v;
      }
    }
#if HAVE_ASYNC_LDS
    if (AMODE == 0) wait_async_lds();
#endif
    __syncthreads();

    // ---- fragments (ISA 16-bit 16x32 layout: lane m=l%16, half=l/16,
    //      K contiguous in groups of 8 at byte offsets half*16 and 32+half*16)
    BF16Frag a[2], b[2];
    #pragma unroll
    for (int ii = 0; ii < 2; ++ii) {
      const int r = (wm * 32 + ii * 16 + lrow) * 32;
      a[ii].h[0] = *(const v8bf*)&sA[r + half * 8];
      a[ii].h[1] = *(const v8bf*)&sA[r + 16 + half * 8];
    }
    #pragma unroll
    for (int jj = 0; jj < 2; ++jj) {
      const int r = (wn * 32 + jj * 16 + lrow) * 32;
      b[jj].h[0] = *(const v8bf*)&sB[r + half * 8];
      b[jj].h[1] = *(const v8bf*)&sB[r + 16 + half * 8];
    }
    #pragma unroll
    for (int ii = 0; ii < 2; ++ii)
      #pragma unroll
      for (int jj = 0; jj < 2; ++jj)
        acc[ii][jj] = __builtin_amdgcn_wmma_f32_16x16x32_bf16(
            false, a[ii].v, false, b[jj].v, (short)0, acc[ii][jj], false, false);
  }

  // ---- epilogue (C/D layout: VGPR v -> row half*8+v, col lrow)
  #pragma unroll
  for (int ii = 0; ii < 2; ++ii) {
    const int rbase = m0 + wm * 32 + ii * 16 + half * 8;
    #pragma unroll
    for (int jj = 0; jj < 2; ++jj) {
      const int gcol = n0 + wn * 32 + jj * 16 + lrow;
      if (gcol < N) {
        #pragma unroll
        for (int v = 0; v < 8; ++v) {
          const int grow = rbase + v;
          if (grow < M) {
            float r = acc[ii][jj][v];
            if (bias)  r += bias[gcol];
            if (bias2) r += bias2[gcol];
            if (BETA)  r += C[(size_t)grow * N + gcol];
            if (ACT == 1) r = fmaxf(r, 0.f);
            C[(size_t)grow * N + gcol] = r;
            if (STORE16) C16[(size_t)grow * N + gcol] = (__bf16)r;
          }
        }
      }
    }
  }
}

// ==================================================================
// Utility / element-wise / edge kernels
// ==================================================================
__global__ void k_init(float* p, long n, float v) {
  const long i = (long)blockIdx.x * blockDim.x + threadIdx.x;
  if (i < n) p[i] = v;
}

__global__ void k_cvt16(const float* __restrict__ in, __bf16* __restrict__ out, long n) {
  const long i = (long)blockIdx.x * blockDim.x + threadIdx.x;
  if (i < n) out[i] = (__bf16)in[i];
}

__device__ __forceinline__ void edge_sd(const int* __restrict__ src, const int* __restrict__ dst,
                                        long e, int E, int& s, int& d) {
  if (e < E) { s = src[e]; d = dst[e]; } else { s = d = (int)(e - E); }
}

// logits[e,h] = sum_d att[h][d]*leakyrelu(xl[src]+xr[dst]); segment-max via atomics
__global__ void k_edge_logits(const int* __restrict__ src, const int* __restrict__ dst,
                              int E, int Nn,
                              const float* __restrict__ xl, const float* __restrict__ xr,
                              const float* __restrict__ att,
                              float* __restrict__ elog, float* __restrict__ maxb)
{
  const long idx = (long)blockIdx.x * blockDim.x + threadIdx.x;
  const long tot = (long)(E + Nn) * NH;
  if (idx >= tot) return;
  const int hh = (int)(idx & 3);
  const long e = idx >> 2;
  int s, d; edge_sd(src, dst, e, E, s, d);
  const float* pl = xl + (size_t)s * HD + hh * DHD;
  const float* pr = xr + (size_t)d * HD + hh * DHD;
  const float* pa = att + hh * DHD;
  float sum = 0.f;
  #pragma unroll 8
  for (int k = 0; k < DHD; ++k) {
    float u = pl[k] + pr[k];
    u = (u > 0.f) ? u : 0.2f * u;
    sum += pa[k] * u;
  }
  elog[idx] = sum;
  atomicMaxF(&maxb[(size_t)d * NH + hh], sum);
}

__global__ void k_edge_exp(const int* __restrict__ src, const int* __restrict__ dst,
                           int E, int Nn,
                           float* __restrict__ elog, const float* __restrict__ maxb,
                           float* __restrict__ den)
{
  const long idx = (long)blockIdx.x * blockDim.x + threadIdx.x;
  const long tot = (long)(E + Nn) * NH;
  if (idx >= tot) return;
  const int hh = (int)(idx & 3);
  const long e = idx >> 2;
  int s, d; edge_sd(src, dst, e, E, s, d);
  const float a = __expf(elog[idx] - maxb[(size_t)d * NH + hh]);
  elog[idx] = a;
  atomicAdd(&den[(size_t)d * NH + hh], a);
}

__global__ void k_edge_scatter(const int* __restrict__ src, const int* __restrict__ dst,
                               int E, int Nn,
                               const float* __restrict__ elog, const float* __restrict__ den,
                               const float* __restrict__ xl, float* __restrict__ accb)
{
  const long idx = (long)blockIdx.x * blockDim.x + threadIdx.x;
  const long tot = (long)(E + Nn) * NH;
  if (idx >= tot) return;
  const int hh = (int)(idx & 3);
  const long e = idx >> 2;
  int s, d; edge_sd(src, dst, e, E, s, d);
  const float alpha = elog[idx] / fmaxf(den[(size_t)d * NH + hh], 1e-16f);
  const float* pl = xl + (size_t)s * HD + hh * DHD;
  float* pd = accb + (size_t)d * HD + hh * DHD;
  #pragma unroll 4
  for (int k = 0; k < DHD; ++k) atomicAdd(&pd[k], alpha * pl[k]);
}

// logits = relu-hidden @ a2W + a2b ; ew = sigmoid(l)*exp(-lam*td) ; entropy sums
__global__ __launch_bounds__(256)
void k_rowdot(const float* __restrict__ z, const float* __restrict__ a2w,
              const float* __restrict__ a2b, const float* __restrict__ td,
              const float* __restrict__ lam, float* __restrict__ ewb,
              float* __restrict__ ST, int E)
{
  const int row  = blockIdx.x * 8 + (threadIdx.x >> 5);
  const int lane = threadIdx.x & 31;
  float sum = 0.f;
  if (row < E) {
    #pragma unroll
    for (int i = 0; i < 4; ++i) {
      const int c = lane + i * 32;
      sum += z[(size_t)row * 128 + c] * a2w[c];
    }
  }
  #pragma unroll
  for (int off = 16; off > 0; off >>= 1) sum += __shfl_xor(sum, off, 32);
  if (row < E && lane == 0) {
    const float l = sum + a2b[0];
    ewb[row] = sigf(l) * __expf(-lam[0] * td[row]);
    const float lc = fminf(fmaxf(l, -10.f), 10.f);
    const float el = __expf(lc);
    atomicAdd(&ST[0], el);        // S = sum e^l
    atomicAdd(&ST[1], lc * el);   // T = sum l e^l
  }
}

__global__ void k_entropy(const float* __restrict__ ST, float* __restrict__ out) {
  const float S = ST[0], T = ST[1];
  const float e = __logf(S) - T / S;   // -sum p log p with p = e^l / S
  out[0] = fminf(fmaxf(e, 0.f), 100.f);
}

__global__ void k_ew_node(const int* __restrict__ dst0, const float* __restrict__ ewb,
                          float* __restrict__ nw, float* __restrict__ nc, int E) {
  const int e = blockIdx.x * blockDim.x + threadIdx.x;
  if (e < E) { atomicAdd(&nw[dst0[e]], ewb[e]); atomicAdd(&nc[dst0[e]], 1.f); }
}
__global__ void k_nwscale(float* __restrict__ nw, const float* __restrict__ nc, int Nn) {
  const int i = blockIdx.x * blockDim.x + threadIdx.x;
  if (i < Nn) nw[i] = nw[i] / fmaxf(nc[i], 1.f);
}

// residual/scale + LayerNorm + ReLU; one wave32 per node (8 ch/lane); dual fp32+bf16 out
__global__ __launch_bounds__(256)
void k_ln(const float* __restrict__ accb, const float* __restrict__ bias,
          const float* __restrict__ gam, const float* __restrict__ bet,
          const float* __restrict__ scale, float* __restrict__ h,
          __bf16* __restrict__ h16, int Nn, int layer0)
{
  const int node = blockIdx.x * 8 + (threadIdx.x >> 5);
  const int lane = threadIdx.x & 31;
  if (node >= Nn) return;
  const size_t base = (size_t)node * HD;
  const float s = layer0 ? scale[node] : 0.f;
  float v[8];
  float mean = 0.f;
  #pragma unroll
  for (int i = 0; i < 8; ++i) {
    const int c = lane + i * 32;
    float t = accb[base + c] + bias[c];
    if (layer0) t *= s; else t += h[base + c];
    v[i] = t; mean += t;
  }
  #pragma unroll
  for (int off = 16; off > 0; off >>= 1) mean += __shfl_xor(mean, off, 32);
  mean *= (1.f / HD);
  float var = 0.f;
  #pragma unroll
  for (int i = 0; i < 8; ++i) { const float d = v[i] - mean; var += d * d; }
  #pragma unroll
  for (int off = 16; off > 0; off >>= 1) var += __shfl_xor(var, off, 32);
  var *= (1.f / HD);
  const float inv = rsqrtf(var + 1e-5f);
  #pragma unroll
  for (int i = 0; i < 8; ++i) {
    const int c = lane + i * 32;
    const float o = fmaxf((v[i] - mean) * inv * gam[c] + bet[c], 0.f);
    h[base + c]   = o;
    h16[base + c] = (__bf16)o;
  }
}

__global__ void k_lstm_act(const float* __restrict__ gi, const float* __restrict__ gg,
                           const float* __restrict__ go,
                           float* __restrict__ outc, float* __restrict__ outl,
                           __bf16* __restrict__ lo16, long n) {
  const long i = (long)blockIdx.x * blockDim.x + threadIdx.x;
  if (i < n) {
    const float c = sigf(gi[i]) * tanhf(gg[i]);
    const float l = sigf(go[i]) * tanhf(c);
    outc[i] = c;
    outl[i] = l;
    lo16[i] = (__bf16)l;
  }
}

__global__ void k_gate(const float* __restrict__ zacc, const float* __restrict__ lo,
                       const float* __restrict__ h, float* __restrict__ emb, long n) {
  const long i = (long)blockIdx.x * blockDim.x + threadIdx.x;
  if (i < n) {
    const float g = sigf(zacc[i]);
    emb[i] = g * lo[i] + (1.f - g) * h[i];
  }
}

// ==================================================================
// Launcher
// ==================================================================
extern "C" void kernel_launch(void* const* d_in, const int* in_sizes, int n_in,
                              void* d_out, int out_size, void* d_ws, size_t ws_size,
                              hipStream_t stream)
{
  const float* x     = (const float*)d_in[0];
  const int*   ei    = (const int*)d_in[1];
  const float* td    = (const float*)d_in[2];
  const float* Wproj = (const float*)d_in[3];
  const float* bproj = (const float*)d_in[4];
  const float* omega = (const float*)d_in[5];
  const float* phi   = (const float*)d_in[6];
  const float* t2vW  = (const float*)d_in[7];
  const float* t2vb  = (const float*)d_in[8];
  const float* a1W   = (const float*)d_in[9];
  const float* a1b   = (const float*)d_in[10];
  const float* a2W   = (const float*)d_in[11];
  const float* a2b   = (const float*)d_in[12];
  const float* lam   = (const float*)d_in[13];
  const float* Wl[3]  = {(const float*)d_in[14], (const float*)d_in[20], (const float*)d_in[26]};
  const float* Wr[3]  = {(const float*)d_in[15], (const float*)d_in[21], (const float*)d_in[27]};
  const float* att[3] = {(const float*)d_in[16], (const float*)d_in[22], (const float*)d_in[28]};
  const float* bia[3] = {(const float*)d_in[17], (const float*)d_in[23], (const float*)d_in[29]};
  const float* lng[3] = {(const float*)d_in[18], (const float*)d_in[24], (const float*)d_in[30]};
  const float* lnb[3] = {(const float*)d_in[19], (const float*)d_in[25], (const float*)d_in[31]};
  const float* Wih   = (const float*)d_in[32];
  const float* bih   = (const float*)d_in[34];
  const float* bhh   = (const float*)d_in[35];
  const float* gW    = (const float*)d_in[36];
  const float* gb    = (const float*)d_in[37];

  const int Nn = in_sizes[0] / HD;   // 50000
  const int E  = in_sizes[2];        // 300000
  const int Et = E + Nn;             // + self loops

  const int* src = ei;
  const int* dst = ei + E;

  const size_t NE = (size_t)Nn * HD;
  const size_t TE = (size_t)E * HD;

  float* out_emb = (float*)d_out;
  float* out_tf  = out_emb + NE;
  float* out_ent = out_tf + TE;
  float* out_lo  = out_ent + 1;
  float* out_c   = out_lo + NE;

  // ---- fp32 workspace
  float* W = (float*)d_ws;
  float* h    = W;  W += NE;
  float* xlb  = W;  W += NE;
  float* xrb  = W;  W += NE;
  float* accb = W;  W += NE;
  float* zbuf = W;  W += (size_t)E * 128;     // reused later: gi/gg/go, then zacc
  float* elog = W;  W += (size_t)Et * NH;
  float* maxb = W;  W += (size_t)Nn * NH;
  float* den  = W;  W += (size_t)Nn * NH;
  float* ewb  = W;  W += E;
  float* nw   = W;  W += Nn;
  float* nc   = W;  W += Nn;
  float* ST   = W;  W += 2;
  W += 8 - (((uintptr_t)W >> 2) & 7);         // 32B-align the bf16 region

  // ---- bf16 activation workspace (halves GEMM-side HBM traffic; async-copyable)
  __bf16* B16 = (__bf16*)W;
  __bf16* x16  = B16;  B16 += NE;
  __bf16* h16  = B16;  B16 += NE;
  __bf16* lo16 = B16;  B16 += NE;
  __bf16* tf16 = B16;  B16 += TE;

  float* gi   = zbuf;
  float* gg   = zbuf + NE;
  float* go   = zbuf + 2 * NE;
  float* zacc = zbuf;                         // gi consumed before gate GEMMs

  dim3 blk(256);
  auto grM = [](int M)  { return (unsigned)((M + 127) / 128); };
  auto gr1 = [](long n) { return (unsigned)((n + 255) / 256); };

  // init accumulators + bf16 input copy
  k_init<<<1, blk, 0, stream>>>(ST, 2, 0.f);
  k_init<<<gr1(Nn), blk, 0, stream>>>(nw, Nn, 0.f);
  k_init<<<gr1(Nn), blk, 0, stream>>>(nc, Nn, 0.f);
  k_cvt16<<<gr1((long)NE), blk, 0, stream>>>(x, x16, (long)NE);

  // 1) h = relu(x @ Wproj + bproj)  (+ bf16 copy for next GEMMs)
  gemm_wmma<0,0,1,0,1><<<dim3(grM(Nn), 4), blk, 0, stream>>>(x16, Wproj, bproj, nullptr, h, h16, Nn, HD, HD, nullptr, nullptr, nullptr);
  // 2) tf = Time2Vec(td) @ t2vW + t2vb  (A computed in-kernel; fp32 -> d_out, bf16 -> tf16)
  gemm_wmma<1,0,0,0,1><<<dim3(grM(E), 4), blk, 0, stream>>>(nullptr, t2vW, t2vb, nullptr, out_tf, tf16, E, HD, HD, td, omega, phi);
  // 3) z = relu(tf @ a1W + a1b)
  gemm_wmma<0,0,1,0,0><<<dim3(grM(E), 2), blk, 0, stream>>>(tf16, a1W, a1b, nullptr, zbuf, nullptr, E, 128, HD, nullptr, nullptr, nullptr);
  // 4) logits -> ew, entropy partial sums; entropy scalar
  k_rowdot<<<(E + 7) / 8, blk, 0, stream>>>(zbuf, a2W, a2b, td, lam, ewb, ST, E);
  k_entropy<<<1, 1, 0, stream>>>(ST, out_ent);
  // 5) per-node mean edge weight (original edges only)
  k_ew_node<<<gr1(E), blk, 0, stream>>>(dst, ewb, nw, nc, E);
  k_nwscale<<<gr1(Nn), blk, 0, stream>>>(nw, nc, Nn);

  // 6) GATv2 layers
  for (int L = 0; L < 3; ++L) {
    gemm_wmma<0,0,0,0,0><<<dim3(grM(Nn), 4), blk, 0, stream>>>(h16, Wl[L], nullptr, nullptr, xlb, nullptr, Nn, HD, HD, nullptr, nullptr, nullptr);
    gemm_wmma<0,0,0,0,0><<<dim3(grM(Nn), 4), blk, 0, stream>>>(h16, Wr[L], nullptr, nullptr, xrb, nullptr, Nn, HD, HD, nullptr, nullptr, nullptr);
    k_init<<<gr1((long)NE), blk, 0, stream>>>(accb, (long)NE, 0.f);
    k_init<<<gr1((long)Nn * NH), blk, 0, stream>>>(den, (long)Nn * NH, 0.f);
    k_init<<<gr1((long)Nn * NH), blk, 0, stream>>>(maxb, (long)Nn * NH, -1e30f);
    const long tot = (long)Et * NH;
    k_edge_logits <<<gr1(tot), blk, 0, stream>>>(src, dst, E, Nn, xlb, xrb, att[L], elog, maxb);
    k_edge_exp    <<<gr1(tot), blk, 0, stream>>>(src, dst, E, Nn, elog, maxb, den);
    k_edge_scatter<<<gr1(tot), blk, 0, stream>>>(src, dst, E, Nn, elog, den, xlb, accb);
    k_ln<<<(Nn + 7) / 8, blk, 0, stream>>>(accb, bia[L], lng[L], lnb[L], nw, h, h16, Nn, (L == 0) ? 1 : 0);
  }

  // 7) LSTM (h0=c0=0 -> forget gate dead): gi/gg/go = h @ Wih_slice^T + bih + bhh
  gemm_wmma<0,1,0,0,0><<<dim3(grM(Nn), 4), blk, 0, stream>>>(h16, Wih,            bih,       bhh,       gi, nullptr, Nn, HD, HD, nullptr, nullptr, nullptr);
  gemm_wmma<0,1,0,0,0><<<dim3(grM(Nn), 4), blk, 0, stream>>>(h16, Wih + 512 * HD, bih + 512, bhh + 512, gg, nullptr, Nn, HD, HD, nullptr, nullptr, nullptr);
  gemm_wmma<0,1,0,0,0><<<dim3(grM(Nn), 4), blk, 0, stream>>>(h16, Wih + 768 * HD, bih + 768, bhh + 768, go, nullptr, Nn, HD, HD, nullptr, nullptr, nullptr);
  k_lstm_act<<<gr1((long)NE), blk, 0, stream>>>(gi, gg, go, out_c, out_lo, lo16, (long)NE);

  // 8) gate = sigmoid([h, lstm_out] @ gate_W + gate_b) split into two K=256 GEMMs
  gemm_wmma<0,0,0,0,0><<<dim3(grM(Nn), 4), blk, 0, stream>>>(h16,  gW,            nullptr, nullptr, zacc, nullptr, Nn, HD, HD, nullptr, nullptr, nullptr);
  gemm_wmma<0,0,0,1,0><<<dim3(grM(Nn), 4), blk, 0, stream>>>(lo16, gW + 256 * HD, gb,      nullptr, zacc, nullptr, Nn, HD, HD, nullptr, nullptr, nullptr);
  k_gate<<<gr1((long)NE), blk, 0, stream>>>(zacc, out_lo, h, out_emb, (long)NE);
}
